// Model_16870631538950
// MI455X (gfx1250) — matrix operand order repaired and checked by reading, not compiled, not run on hardware
//
#include <hip/hip_runtime.h>
#include <math.h>

typedef _Float16 v16h __attribute__((ext_vector_type(16)));
typedef _Float16 v8h  __attribute__((ext_vector_type(8)));
typedef _Float16 v4h  __attribute__((ext_vector_type(4)));
typedef float    v8f  __attribute__((ext_vector_type(8)));
typedef float    v4f  __attribute__((ext_vector_type(4)));

#define NB   64
#define LSEQ 512
#define MAK  25
#define NH   8
#define NE   64

// ---------------------------------------------------------------------------
// FAST tiled WMMA GEMM: requires M%128==0, N%128==0, K%32==0, all in-bounds.
// C[M,N] (+)= act( A[M,K] * B[K,N] + bias ).  A rows circularly time-shifted
// via (Ltime, tshift).  Register double-buffered staging, hoisted addresses.
// ---------------------------------------------------------------------------
__global__ __launch_bounds__(256) void gemm_fast(
    const float* __restrict__ A, const float* __restrict__ Bm,
    const float* __restrict__ bias, float* __restrict__ C,
    int M, int N, int K, int ldb, int Ltime, int tshift, int accflag, int gelu)
{
  __shared__ _Float16 As[128][48] __attribute__((aligned(16)));
  __shared__ _Float16 Bs[128][48] __attribute__((aligned(16)));
  __shared__ int rmap[128];
  const int tid = threadIdx.x;
  const int m0 = blockIdx.y * 128, n0 = blockIdx.x * 128;
  if (tid < 128) {
    int m = m0 + tid;
    int b = m / Ltime, t = m - b * Ltime;
    int tt = t + tshift; if (tt < 0) tt += Ltime; if (tt >= Ltime) tt -= Ltime;
    rmap[tid] = b * Ltime + tt;
  }
  __syncthreads();
  const int lane = tid & 31, wave = tid >> 5;
  const int wm = wave & 3, wn = wave >> 2;
  const int lrow = lane & 15, lhi = lane >> 4;

  // ---- hoisted staging addresses (loop-invariant per thread) ----
  const int r0  = tid >> 3;          // A row group base (0..31)
  const int kq4 = (tid & 7) * 4;     // A k-quad offset
  const int n4  = (tid & 31) * 4;    // B col quad
  const int kkB = tid >> 5;          // B k base (0..7)
  const float* aPtr[4]; const float* bPtr[4];
  _Float16* asPtr[4]; _Float16* bsPtr[4];
  #pragma unroll
  for (int j = 0; j < 4; j++) {
    int r = r0 + j * 32;
    aPtr[j]  = A + (long)rmap[r] * K + kq4;
    asPtr[j] = &As[r][kq4];
    int kk = kkB + j * 8;
    bPtr[j]  = Bm + (long)kk * ldb + n0 + n4;
    bsPtr[j] = &Bs[n4][kk];
  }

  const v8f vzero = {0.f,0.f,0.f,0.f,0.f,0.f,0.f,0.f};
  v8f acc[2][4];
  #pragma unroll
  for (int i = 0; i < 2; i++)
    #pragma unroll
    for (int j = 0; j < 4; j++) acc[i][j] = vzero;

  // ---- prime first tile ----
  v4f aV[4], bV[4];
  #pragma unroll
  for (int j = 0; j < 4; j++) aV[j] = *reinterpret_cast<const v4f*>(aPtr[j]);
  #pragma unroll
  for (int j = 0; j < 4; j++) bV[j] = *reinterpret_cast<const v4f*>(bPtr[j]);

  const int nIter = K >> 5;
  for (int it = 0; ; ++it) {
    // stage registers -> LDS
    #pragma unroll
    for (int j = 0; j < 4; j++) {
      v4h h = { (_Float16)aV[j].x, (_Float16)aV[j].y, (_Float16)aV[j].z, (_Float16)aV[j].w };
      *reinterpret_cast<v4h*>(asPtr[j]) = h;
    }
    #pragma unroll
    for (int j = 0; j < 4; j++) {
      _Float16* p = bsPtr[j];
      p[0]   = (_Float16)bV[j].x;
      p[48]  = (_Float16)bV[j].y;
      p[96]  = (_Float16)bV[j].z;
      p[144] = (_Float16)bV[j].w;
    }
    __syncthreads();
    // prefetch next tile (overlaps with wmma below)
    if (it + 1 < nIter) {
      #pragma unroll
      for (int j = 0; j < 4; j++) { aPtr[j] += 32; aV[j] = *reinterpret_cast<const v4f*>(aPtr[j]); }
      #pragma unroll
      for (int j = 0; j < 4; j++) { bPtr[j] += (long)32 * ldb; bV[j] = *reinterpret_cast<const v4f*>(bPtr[j]); }
    }
    // fragments + wmma
    v16h afr[2], bfr[4];
    #pragma unroll
    for (int tm = 0; tm < 2; tm++) {
      int r = wm*32 + tm*16 + lrow;
      v8h lo = *reinterpret_cast<const v8h*>(&As[r][lhi*8]);
      v8h hi = *reinterpret_cast<const v8h*>(&As[r][lhi*8 + 16]);
      afr[tm] = __builtin_shufflevector(lo, hi, 0,1,2,3,4,5,6,7,8,9,10,11,12,13,14,15);
    }
    #pragma unroll
    for (int tn = 0; tn < 4; tn++) {
      int c = wn*64 + tn*16 + lrow;
      v8h lo = *reinterpret_cast<const v8h*>(&Bs[c][lhi*8]);
      v8h hi = *reinterpret_cast<const v8h*>(&Bs[c][lhi*8 + 16]);
      bfr[tn] = __builtin_shufflevector(lo, hi, 0,1,2,3,4,5,6,7,8,9,10,11,12,13,14,15);
    }
    #pragma unroll
    for (int tm = 0; tm < 2; tm++)
      #pragma unroll
      for (int tn = 0; tn < 4; tn++)
        acc[tm][tn] = __builtin_amdgcn_wmma_f32_16x16x32_f16(
            false, afr[tm], false, bfr[tn], (short)0, acc[tm][tn], false, false);
    __syncthreads();
    if (it + 1 >= nIter) break;
  }
  // ---- epilogue (no guards: tiles full) ----
  #pragma unroll
  for (int tm = 0; tm < 2; tm++)
    #pragma unroll
    for (int tn = 0; tn < 4; tn++)
      #pragma unroll
      for (int r = 0; r < 8; r++) {
        int mm = m0 + wm*32 + tm*16 + lhi*8 + r;
        int nn = n0 + wn*64 + tn*16 + lrow;
        float v = acc[tm][tn][r];
        if (accflag) v += C[(long)mm * N + nn];
        if (bias)    v += bias[nn];
        if (gelu)    v = 0.5f * v * (1.0f + erff(v * 0.70710678118654752f));
        C[(long)mm * N + nn] = v;
      }
}

// ---------------------------------------------------------------------------
// Generic guarded WMMA GEMM (edge shapes: K=12 embeddings, 64x27 classifier)
// ---------------------------------------------------------------------------
__global__ __launch_bounds__(256) void gemm_f16w(
    const float* __restrict__ A, const float* __restrict__ Bm,
    const float* __restrict__ bias, float* __restrict__ C,
    int M, int N, int K, int ldb, int Ltime, int tshift, int accflag, int gelu)
{
  __shared__ _Float16 As[128][48] __attribute__((aligned(16)));
  __shared__ _Float16 Bs[128][48] __attribute__((aligned(16)));
  __shared__ int rmap[128];
  const int tid = threadIdx.x;
  const int m0 = blockIdx.y * 128, n0 = blockIdx.x * 128;
  if (tid < 128) {
    int m = m0 + tid; int r = -1;
    if (m < M) {
      int b = m / Ltime, t = m - b * Ltime;
      int tt = t + tshift; if (tt < 0) tt += Ltime; if (tt >= Ltime) tt -= Ltime;
      r = b * Ltime + tt;
    }
    rmap[tid] = r;
  }
  __syncthreads();
  const int lane = tid & 31, wave = tid >> 5;
  const int wm = wave & 3, wn = wave >> 2;
  const int lrow = lane & 15, lhi = lane >> 4;
  const v8f vzero = {0.f,0.f,0.f,0.f,0.f,0.f,0.f,0.f};
  v8f acc[2][4];
  #pragma unroll
  for (int i = 0; i < 2; i++)
    #pragma unroll
    for (int j = 0; j < 4; j++) acc[i][j] = vzero;

  for (int k0 = 0; k0 < K; k0 += 32) {
    for (int i = tid; i < 128 * 32; i += 256) {
      int r = i >> 5, kk = i & 31;
      float v = 0.f; int gr = rmap[r]; int gk = k0 + kk;
      if (gr >= 0 && gk < K) v = A[(long)gr * K + gk];
      As[r][kk] = (_Float16)v;
    }
    for (int i = tid; i < 32 * 128; i += 256) {
      int kk = i >> 7, n = i & 127;
      float v = 0.f; int gk = k0 + kk, gn = n0 + n;
      if (gk < K && gn < N) v = Bm[(long)gk * ldb + gn];
      Bs[n][kk] = (_Float16)v;
    }
    __syncthreads();
    v16h afr[2], bfr[4];
    #pragma unroll
    for (int tm = 0; tm < 2; tm++) {
      int r = wm*32 + tm*16 + lrow;
      v8h lo = *reinterpret_cast<const v8h*>(&As[r][lhi*8]);
      v8h hi = *reinterpret_cast<const v8h*>(&As[r][lhi*8 + 16]);
      afr[tm] = __builtin_shufflevector(lo, hi, 0,1,2,3,4,5,6,7,8,9,10,11,12,13,14,15);
    }
    #pragma unroll
    for (int tn = 0; tn < 4; tn++) {
      int c = wn*64 + tn*16 + lrow;
      v8h lo = *reinterpret_cast<const v8h*>(&Bs[c][lhi*8]);
      v8h hi = *reinterpret_cast<const v8h*>(&Bs[c][lhi*8 + 16]);
      bfr[tn] = __builtin_shufflevector(lo, hi, 0,1,2,3,4,5,6,7,8,9,10,11,12,13,14,15);
    }
    #pragma unroll
    for (int tm = 0; tm < 2; tm++)
      #pragma unroll
      for (int tn = 0; tn < 4; tn++)
        acc[tm][tn] = __builtin_amdgcn_wmma_f32_16x16x32_f16(
            false, afr[tm], false, bfr[tn], (short)0, acc[tm][tn], false, false);
    __syncthreads();
  }
  #pragma unroll
  for (int tm = 0; tm < 2; tm++)
    #pragma unroll
    for (int tn = 0; tn < 4; tn++)
      #pragma unroll
      for (int r = 0; r < 8; r++) {
        int mm = m0 + wm*32 + tm*16 + lhi*8 + r;
        int nn = n0 + wn*64 + tn*16 + lrow;
        if (mm < M && nn < N) {
          float v = acc[tm][tn][r];
          if (accflag) v += C[(long)mm * N + nn];
          if (bias)    v += bias[nn];
          if (gelu)    v = 0.5f * v * (1.0f + erff(v * 0.70710678118654752f));
          C[(long)mm * N + nn] = v;
        }
      }
}

// ----------------- DFT basis: F[512][128] (cos | -sin), I[128][512] ---------
__global__ void fill_dft_kernel(float* __restrict__ F, float* __restrict__ Iv) {
  int tid = blockIdx.x * 256 + threadIdx.x;
  if (tid >= 512 * 128) return;
  int t = tid >> 7, j = tid & 127, mj = j & 63;
  double th = (2.0 * 3.14159265358979323846 * (double)(mj * t)) / 512.0;
  double c = cos(th), s = sin(th);
  F[t * 128 + j] = (j < 64) ? (float)c : (float)(-s);
  float iv;
  if (j < 64) iv = (float)(((j == 0) ? 1.0 : 2.0 * c) / 512.0);
  else        iv = (mj == 0) ? 0.f : (float)(-2.0 * s / 512.0);
  Iv[j * 512 + t] = iv;
}

// ----------------- series_decomp: s = z - MA25(z), z = x (+ y) -------------
__global__ void decomp_kernel(const float* __restrict__ x, const float* __restrict__ y,
                              float* __restrict__ s_out, float* __restrict__ t_out,
                              int L, int C, int total, int t_mode) {
  int tid = blockIdx.x * 256 + threadIdx.x;
  if (tid >= total) return;
  int c = tid % C, t = (tid / C) % L, b = tid / (C * L);
  const int pad = (MAK - 1) / 2;
  float sum = 0.f;
  for (int d = -pad; d <= pad; d++) {
    int tt = t + d; tt = tt < 0 ? 0 : (tt >= L ? L - 1 : tt);
    long idx = ((long)(b * L + tt)) * C + c;
    float v = x[idx]; if (y) v += y[idx];
    sum += v;
  }
  float ma = sum * (1.0f / MAK);
  long ci = ((long)(b * L + t)) * C + c;
  float z = x[ci]; if (y) z += y[ci];
  s_out[ci] = z - ma;
  if (t_mode == 1) t_out[ci] = ma;
  else if (t_mode == 2) t_out[ci] += ma;
}

// ----------------- misc small kernels --------------------------------------
__global__ void meanx_kernel(const float* __restrict__ x, float* __restrict__ mx) {
  int tid = blockIdx.x * 256 + threadIdx.x;
  if (tid >= NB * 12) return;
  int b = tid / 12, c = tid % 12;
  float s = 0.f;
  for (int t = 0; t < LSEQ; t++) s += x[((long)(b * LSEQ + t)) * 12 + c];
  mx[tid] = s * (1.0f / LSEQ);
}

__global__ void build_init_kernel(const float* __restrict__ sfull, const float* __restrict__ tfull,
                                  const float* __restrict__ mx,
                                  float* __restrict__ sinit, float* __restrict__ tinit) {
  int tid = blockIdx.x * 256 + threadIdx.x;
  if (tid >= NB * 512 * 12) return;
  int c = tid % 12, t = (tid / 12) % 512, b = tid / (512 * 12);
  float s, tr;
  if (t < 256) {
    long src = ((long)(b * LSEQ + 256 + t)) * 12 + c;
    s = sfull[src]; tr = tfull[src];
  } else { s = 0.f; tr = mx[b * 12 + c]; }
  sinit[tid] = s; tinit[tid] = tr;
}

// per-b 512x512 transpose: out[b][d][t] = in[b][t][d]
__global__ void transpose512(const float* __restrict__ in, float* __restrict__ out) {
  __shared__ float tile[32][33];
  int b = blockIdx.z;
  int xb = blockIdx.x * 32, yb = blockIdx.y * 32;
  const float* ib = in + (long)b * 262144;
  float* ob = out + (long)b * 262144;
  for (int i = 0; i < 32; i += 8)
    tile[threadIdx.y + i][threadIdx.x] = ib[(long)(yb + threadIdx.y + i) * 512 + xb + threadIdx.x];
  __syncthreads();
  for (int i = 0; i < 32; i += 8)
    ob[(long)(xb + threadIdx.y + i) * 512 + yb + threadIdx.x] = tile[threadIdx.x][threadIdx.y + i];
}

// out[(b,h,o),m] = scale * sum_i sel[(b,h,i),m] * (wr+j*wi)[h,i,o,m]
__global__ void modemix_kernel(const float* __restrict__ S, const float* __restrict__ WR,
                               const float* __restrict__ WI, float* __restrict__ O, float scale) {
  int tid = blockIdx.x * 256 + threadIdx.x;
  int m = tid & 63, o = (tid >> 6) & 63, h = (tid >> 12) & 7, b = tid >> 15;
  int bh = b * 8 + h;
  float re = 0.f, im = 0.f;
  for (int i = 0; i < 64; i++) {
    long sr = ((long)(bh * 64 + i)) * 128;
    float s_r = S[sr + m], s_i = S[sr + 64 + m];
    long wi_ = ((long)((h * 64 + i) * 64 + o)) * 64 + m;
    float w_r = WR[wi_], w_i = WI[wi_];
    re += s_r * w_r - s_i * w_i;
    im += s_r * w_i + s_i * w_r;
  }
  long oi = ((long)(bh * 64 + o)) * 128;
  O[oi + m] = re * scale; O[oi + 64 + m] = im * scale;
}

// xqk[b,h,x,y] = ctanh( sum_e q[b,h,e,x] * k[b,h,e,y] )
__global__ void xqk_tanh_kernel(const float* __restrict__ Q, const float* __restrict__ Kf,
                                float* __restrict__ O) {
  int tid = blockIdx.x * 256 + threadIdx.x;
  int y = tid & 63, x = (tid >> 6) & 63, h = (tid >> 12) & 7, b = tid >> 15;
  int bh = b * 8 + h;
  float re = 0.f, im = 0.f;
  for (int e = 0; e < 64; e++) {
    long r = ((long)(bh * 64 + e)) * 128;
    float qr = Q[r + x], qi = Q[r + 64 + x];
    float kr = Kf[r + y], ki = Kf[r + 64 + y];
    re += qr * kr - qi * ki;
    im += qr * ki + qi * kr;
  }
  float a2 = 2.f * re; a2 = a2 > 30.f ? 30.f : (a2 < -30.f ? -30.f : a2);
  float b2 = 2.f * im;
  float ex = __expf(a2), exi = 1.f / ex;
  float ch = 0.5f * (ex + exi), sh = 0.5f * (ex - exi);
  float den = ch + __cosf(b2);
  long oi = ((long)(bh * 64 + x)) * 128;
  O[oi + y] = sh / den; O[oi + 64 + y] = __sinf(b2) / den;
}

// xqkv[(b,h,e),x] = sum_y xqk[b,h,x,y] * k[b,h,e,y]
__global__ void xqkv_kernel(const float* __restrict__ T, const float* __restrict__ Kf,
                            float* __restrict__ O) {
  int tid = blockIdx.x * 256 + threadIdx.x;
  int x = tid & 63, e = (tid >> 6) & 63, h = (tid >> 12) & 7, b = tid >> 15;
  int bh = b * 8 + h;
  long tr = ((long)(bh * 64 + x)) * 128;
  long kr = ((long)(bh * 64 + e)) * 128;
  float re = 0.f, im = 0.f;
  for (int y = 0; y < 64; y++) {
    float t_r = T[tr + y], t_i = T[tr + 64 + y];
    float k_r = Kf[kr + y], k_i = Kf[kr + 64 + y];
    re += t_r * k_r - t_i * k_i;
    im += t_r * k_i + t_i * k_r;
  }
  long oi = ((long)(bh * 64 + e)) * 128;
  O[oi + x] = re; O[oi + 64 + x] = im;
}

// LayerNorm over channels (D=512), one block per (b,t) row
__global__ void ln_row_kernel(const float* __restrict__ x, const float* __restrict__ w,
                              const float* __restrict__ bb, float* __restrict__ out) {
  __shared__ float s1[256], s2[256];
  int row = blockIdx.x, tid = threadIdx.x;
  long base = (long)row * 512;
  float v0 = x[base + tid], v1 = x[base + 256 + tid];
  s1[tid] = v0 + v1; s2[tid] = v0 * v0 + v1 * v1;
  __syncthreads();
  for (int s = 128; s > 0; s >>= 1) {
    if (tid < s) { s1[tid] += s1[tid + s]; s2[tid] += s2[tid + s]; }
    __syncthreads();
  }
  float mu = s1[0] * (1.f / 512.f);
  float var = s2[0] * (1.f / 512.f) - mu * mu;
  float rstd = rsqrtf(var + 1e-5f);
  out[base + tid]       = (v0 - mu) * rstd * w[tid]       + bb[tid];
  out[base + 256 + tid] = (v1 - mu) * rstd * w[tid + 256] + bb[tid + 256];
}

__global__ void colmean_kernel(const float* __restrict__ x, float* __restrict__ cm) {
  int tid = blockIdx.x * 256 + threadIdx.x;
  if (tid >= NB * 512) return;
  int b = tid >> 9, d = tid & 511;
  float s = 0.f;
  for (int t = 0; t < 512; t++) s += x[((long)(b * 512 + t)) * 512 + d];
  cm[tid] = s * (1.f / 512.f);
}

__global__ void submean_kernel(const float* __restrict__ x, const float* __restrict__ cm,
                               float* __restrict__ out) {
  long tid = (long)blockIdx.x * 256 + threadIdx.x;
  if (tid >= (long)NB * 512 * 512) return;
  int d = (int)(tid & 511); int b = (int)(tid >> 18);
  out[tid] = x[tid] - cm[b * 512 + d];
}

__global__ void pool_kernel(const float* __restrict__ a, const float* __restrict__ b2,
                            float* __restrict__ out) {
  int tid = blockIdx.x * 256 + threadIdx.x;
  if (tid >= NB * 512) return;
  int b = tid >> 9, d = tid & 511;
  float s = 0.f;
  for (int t = 0; t < 512; t++) {
    long i = ((long)(b * 512 + t)) * 512 + d;
    s += a[i] + b2[i];
  }
  out[tid] = s * (1.f / 512.f);
}

// ---------------------------------------------------------------------------
extern "C" void kernel_launch(void* const* d_in, const int* in_sizes, int n_in,
                              void* d_out, int out_size, void* d_ws, size_t ws_size,
                              hipStream_t stream) {
  (void)in_sizes; (void)n_in; (void)out_size; (void)ws_size;
  const float* x_enc     = (const float*)d_in[0];
  const float* enc_emb_w = (const float*)d_in[1];
  const float* dec_emb_w = (const float*)d_in[2];
  const float* trend_w   = (const float*)d_in[3];
  const float* trend_b   = (const float*)d_in[4];
  const float* cls_w     = (const float*)d_in[5];
  const float* cls_b     = (const float*)d_in[6];
  const float* enc_ln_w  = (const float*)d_in[7];
  const float* enc_ln_b  = (const float*)d_in[8];
  const float* dec_ln_w  = (const float*)d_in[9];
  const float* dec_ln_b  = (const float*)d_in[10];

  const size_t SZ = (size_t)NB * 512 * 512;
  float* W = (float*)d_ws;
  float* ENC   = W;            float* DECa = W + SZ;     float* DECb = W + 2*SZ;
  float* TREND = W + 3*SZ;     float* TSUM = W + 4*SZ;
  float* SCRa  = W + 5*SZ;     float* SCRb = W + 6*SZ;   float* SCRc = W + 7*SZ;
  float* FFNH  = W + 8*SZ;
  size_t cur = 9*SZ;
  const size_t SELSZ = (size_t)NB * NH * NE * 128;
  float* SELQ = W + cur; cur += SELSZ;
  float* SELK = W + cur; cur += SELSZ;
  float* SELO = W + cur; cur += SELSZ;
  float* XQK  = W + cur; cur += SELSZ;
  float* DFTF = W + cur; cur += 512*128;
  float* DFTI = W + cur; cur += 128*512;
  float* SFULL = W + cur; cur += (size_t)NB*512*12;
  float* TFULL = W + cur; cur += (size_t)NB*512*12;
  float* SINIT = W + cur; cur += (size_t)NB*512*12;
  float* TINIT = W + cur; cur += (size_t)NB*512*12;
  float* MEANX = W + cur; cur += 1024;
  float* CMEAN = W + cur; cur += (size_t)NB*512;
  float* POOL  = W + cur; cur += (size_t)NB*512;

  const int M = NB * 512;   // 32768
  auto gemm = [&](const float* A, const float* B, const float* bias, float* C,
                  int m, int n, int k, int ldb, int lt, int sh, int acc, int gl) {
    dim3 g((n + 127) / 128, (m + 127) / 128);
    if (((m & 127) == 0) && ((n & 127) == 0) && ((k & 31) == 0))
      gemm_fast<<<g, 256, 0, stream>>>(A, B, bias, C, m, n, k, ldb, lt, sh, acc, gl);
    else
      gemm_f16w<<<g, 256, 0, stream>>>(A, B, bias, C, m, n, k, ldb, lt, sh, acc, gl);
  };
  auto blocks = [](long n) { return dim3((unsigned)((n + 255) / 256)); };

  // ---- constants / input prep ----
  fill_dft_kernel<<<blocks(512*128), 256, 0, stream>>>(DFTF, DFTI);
  meanx_kernel<<<blocks(NB*12), 256, 0, stream>>>(x_enc, MEANX);
  decomp_kernel<<<blocks((long)NB*512*12), 256, 0, stream>>>(x_enc, nullptr, SFULL, TFULL, 512, 12, NB*512*12, 1);
  build_init_kernel<<<blocks((long)NB*512*12), 256, 0, stream>>>(SFULL, TFULL, MEANX, SINIT, TINIT);
  gemm(TINIT, trend_w, trend_b, TREND, M, 512, 12, 512, 512, 0, 0, 0);
  for (int j = 0; j < 3; j++)
    gemm(x_enc, enc_emb_w + (size_t)j*12*512, nullptr, ENC, M, 512, 12, 512, 512, j-1, j>0, 0);

  auto fourier_self = [&](const float* X, const float* qw, const float* qb,
                          const float* fr, const float* fi,
                          const float* ow, const float* ob, float* out) {
    gemm(X, qw, qb, SCRa, M, 512, 512, 512, 512, 0, 0, 0);
    transpose512<<<dim3(16,16,NB), dim3(32,8), 0, stream>>>(SCRa, SCRb);
    gemm(SCRb, DFTF, nullptr, SELQ, M, 128, 512, 128, 512, 0, 0, 0);
    modemix_kernel<<<blocks((long)NB*8*64*64), 256, 0, stream>>>(SELQ, fr, fi, SELO, 1.0f);
    gemm(SELO, DFTI, nullptr, SCRb, M, 512, 128, 512, 512, 0, 0, 0);
    gemm(SCRb, ow, ob, out, M, 512, 512, 512, 512, 0, 0, 0);
  };
  auto ffn = [&](const float* X, const float* c1, const float* c2, float* Y) {
    for (int j = 0; j < 4; j++) {
      gemm(X, c1 + (size_t)j*512, nullptr, FFNH, M, 512, 512, 2048, 512, 0, 0, 1);
      gemm(FFNH, c2 + (size_t)j*512*512, nullptr, Y, M, 512, 512, 512, 512, 0, j>0, 0);
    }
  };

  // ---- encoder layers ----
  for (int l = 0; l < 2; l++) {
    int base = 11 + l * 12;
    const float* qw = (const float*)d_in[base+0]; const float* qb = (const float*)d_in[base+1];
    const float* ow = (const float*)d_in[base+6]; const float* ob = (const float*)d_in[base+7];
    const float* fr = (const float*)d_in[base+8]; const float* fi = (const float*)d_in[base+9];
    const float* c1 = (const float*)d_in[base+10]; const float* c2 = (const float*)d_in[base+11];
    fourier_self(ENC, qw, qb, fr, fi, ow, ob, SCRa);
    decomp_kernel<<<blocks((long)M*512), 256, 0, stream>>>(ENC, SCRa, SCRc, nullptr, 512, 512, M*512, 0);
    ffn(SCRc, c1, c2, SCRa);
    decomp_kernel<<<blocks((long)M*512), 256, 0, stream>>>(SCRc, SCRa, ENC, nullptr, 512, 512, M*512, 0);
  }
  ln_row_kernel<<<M, 256, 0, stream>>>(ENC, enc_ln_w, enc_ln_b, SCRa);
  colmean_kernel<<<blocks(NB*512), 256, 0, stream>>>(SCRa, CMEAN);
  submean_kernel<<<blocks((long)M*512), 256, 0, stream>>>(SCRa, CMEAN, ENC);

  // ---- decoder embedding ----
  for (int j = 0; j < 3; j++)
    gemm(SINIT, dec_emb_w + (size_t)j*12*512, nullptr, DECa, M, 512, 12, 512, 512, j-1, j>0, 0);

  // ---- decoder layer ----
  {
    int sb = 35, cb = 45;
    const float* sqw = (const float*)d_in[sb+0]; const float* sqb = (const float*)d_in[sb+1];
    const float* sow = (const float*)d_in[sb+6]; const float* sob = (const float*)d_in[sb+7];
    const float* sfr = (const float*)d_in[sb+8]; const float* sfi = (const float*)d_in[sb+9];
    const float* cqw = (const float*)d_in[cb+0]; const float* cqb = (const float*)d_in[cb+1];
    const float* ckw = (const float*)d_in[cb+2]; const float* ckb = (const float*)d_in[cb+3];
    const float* cow = (const float*)d_in[cb+6]; const float* cob = (const float*)d_in[cb+7];
    const float* cfr = (const float*)d_in[cb+8]; const float* cfi = (const float*)d_in[cb+9];
    const float* c1  = (const float*)d_in[55];   const float* c2  = (const float*)d_in[56];
    const float* pw  = (const float*)d_in[57];

    fourier_self(DECa, sqw, sqb, sfr, sfi, sow, sob, SCRa);
    decomp_kernel<<<blocks((long)M*512), 256, 0, stream>>>(DECa, SCRa, DECb, TSUM, 512, 512, M*512, 1);
    gemm(DECb, cqw, cqb, SCRa, M, 512, 512, 512, 512, 0, 0, 0);
    transpose512<<<dim3(16,16,NB), dim3(32,8), 0, stream>>>(SCRa, SCRb);
    gemm(SCRb, DFTF, nullptr, SELQ, M, 128, 512, 128, 512, 0, 0, 0);
    gemm(ENC, ckw, ckb, SCRa, M, 512, 512, 512, 512, 0, 0, 0);
    transpose512<<<dim3(16,16,NB), dim3(32,8), 0, stream>>>(SCRa, SCRb);
    gemm(SCRb, DFTF, nullptr, SELK, M, 128, 512, 128, 512, 0, 0, 0);
    xqk_tanh_kernel<<<blocks((long)NB*8*64*64), 256, 0, stream>>>(SELQ, SELK, XQK);
    xqkv_kernel<<<blocks((long)NB*8*64*64), 256, 0, stream>>>(XQK, SELK, SELO);
    modemix_kernel<<<blocks((long)NB*8*64*64), 256, 0, stream>>>(SELO, cfr, cfi, SELQ,
                                                                  1.0f / (512.f * 512.f));
    gemm(SELQ, DFTI, nullptr, SCRb, M, 512, 128, 512, 512, 0, 0, 0);
    gemm(SCRb, cow, cob, SCRa, M, 512, 512, 512, 512, 0, 0, 0);
    decomp_kernel<<<blocks((long)M*512), 256, 0, stream>>>(DECb, SCRa, DECa, TSUM, 512, 512, M*512, 2);
    ffn(DECa, c1, c2, SCRa);
    decomp_kernel<<<blocks((long)M*512), 256, 0, stream>>>(DECa, SCRa, DECb, TSUM, 512, 512, M*512, 2);
    for (int j = 0; j < 3; j++)
      gemm(TSUM, pw + (size_t)j*512*512, nullptr, TREND, M, 512, 512, 512, 512, j-1, 1, 0);
  }

  // ---- decoder my_layernorm -> seasonal_part (DECa) ----
  ln_row_kernel<<<M, 256, 0, stream>>>(DECb, dec_ln_w, dec_ln_b, SCRa);
  colmean_kernel<<<blocks(NB*512), 256, 0, stream>>>(SCRa, CMEAN);
  submean_kernel<<<blocks((long)M*512), 256, 0, stream>>>(SCRa, CMEAN, DECa);

  // ---- pool + classifier ----
  pool_kernel<<<blocks(NB*512), 256, 0, stream>>>(TREND, DECa, POOL);
  gemm(POOL, cls_w, cls_b, (float*)d_out, NB, 27, 512, 27, 512, 0, 0, 0);
}
